// DCTAttentionIdeal_71038759076393
// MI455X (gfx1250) — matrix operand order repaired and checked by reading, not compiled, not run on hardware
//
#include <hip/hip_runtime.h>
#include <hip/hip_bf16.h>

// DCT attention restructured as: out = D^T (D (softmax(QsKs^T) @ (D^T (D Vm)))),
// i.e. flash attention with value matrix U = D^T D Vm, then DCT-project the
// output. All GEMMs run on v_wmma_f32_16x16x32_f16 (f16 operands, f32 acc).
// Flash kernel uses a 32-query-row tile per wave so every global B-fragment
// (K and U) feeds two WMMAs -> ~32 flop/byte of L2 traffic.

#define BB 2
#define HH 16
#define NN 2048
#define DD 64
#define MM 256

typedef __attribute__((ext_vector_type(16))) _Float16 v16h;
typedef __attribute__((ext_vector_type(8)))  _Float16 v8h;
typedef __attribute__((ext_vector_type(8)))  float    v8f;

// ---- WMMA helpers -----------------------------------------------------------

__device__ __forceinline__ v8f wmma32(v16h a, v16h b, v8f c) {
  return __builtin_amdgcn_wmma_f32_16x16x32_f16(
      /*neg_a=*/false, a, /*neg_b=*/false, b,
      /*c_mod=*/(short)0, c, /*reuse_a=*/false, /*reuse_b=*/false);
}

// A-operand fragment (16 rows x 32 K) from row-major f16, leading dim ld.
// Lane L holds row L&15; halves 0..7 -> K=(L>>4)*8+h, halves 8..15 -> K=16+(L>>4)*8+h.
__device__ __forceinline__ v16h load_frag_a(const _Float16* src, int ld) {
  int lane = threadIdx.x & 31;
  int r  = lane & 15;
  int kb = (lane >> 4) * 8;
  const _Float16* p = src + (long)r * ld + kb;
  v8h lo = *(const v8h*)(p);
  v8h hi = *(const v8h*)(p + 16);
  v16h f;
#pragma unroll
  for (int i = 0; i < 8; ++i) { f[i] = lo[i]; f[8 + i] = hi[i]; }
  return f;
}

// B-operand fragment (32 K x 16 cols), memory holds B^T row-major (Bt[col][k]).
// Lane L holds col L&15; halves 0..7 -> K=(L>>4)*8+h, halves 8..15 -> K=16+(L>>4)*8+h.
__device__ __forceinline__ v16h load_frag_b(const _Float16* bt, int ld) {
  int lane = threadIdx.x & 31;
  int c  = lane & 15;
  int kb = (lane >> 4) * 8;
  const _Float16* p = bt + (long)c * ld + kb;
  v8h lo = *(const v8h*)(p);
  v8h hi = *(const v8h*)(p + 16);
  v16h f;
#pragma unroll
  for (int i = 0; i < 8; ++i) { f[i] = lo[i]; f[8 + i] = hi[i]; }
  return f;
}

// ---- prep kernels -----------------------------------------------------------

// Kh = f16(K * scale * mask[b,n]) kept [n,d]-major (that IS the B^T layout for S).
// Vmt = f16(V * mask) stored transposed [d,n] per head (B^T layout for D@Vm).
__global__ void prep_kv_kernel(const float* __restrict__ K,
                               const float* __restrict__ V,
                               const float* __restrict__ mask,
                               _Float16* __restrict__ Kh,
                               _Float16* __restrict__ Vmt) {
  long idx = (long)blockIdx.x * blockDim.x + threadIdx.x;  // over B*H*N*D
  int d   = (int)(idx & (DD - 1));
  long r  = idx >> 6;
  int n   = (int)(r & (NN - 1));
  long hh = r >> 11;                 // 0..31
  int b   = (int)(hh >> 4);
  float m = mask[(long)b * NN + n];
  const float scale = 0.35355339059327373f;  // 64^(-1/4)
  Kh[idx] = (_Float16)(K[idx] * scale * m);
  Vmt[(hh * DD + d) * NN + n] = (_Float16)(V[idx] * m);
}

// Dh = f16(Q_dct) [M,N]; Dth = transpose [N,M].
__global__ void prep_dct_kernel(const float* __restrict__ Qdct,
                                _Float16* __restrict__ Dh,
                                _Float16* __restrict__ Dth) {
  long idx = (long)blockIdx.x * blockDim.x + threadIdx.x;  // over M*N
  int n = (int)(idx & (NN - 1));
  int m = (int)(idx >> 11);
  float v = Qdct[idx];
  Dh[idx] = (_Float16)v;
  Dth[(long)n * MM + m] = (_Float16)v;
}

// ---- generic per-head 16x16-tile GEMM, f16 transposed output ----------------
// C(mt*16.., dt*16..) = A[mt rows] @ Bt^T, K-loop in steps of 32.
// blockIdx.x = mt*4 + dt (output always has 64 columns -> 4 col-tiles).
__global__ void gemm_h16_kernel(const _Float16* __restrict__ A, long strideA, int lda,
                                const _Float16* __restrict__ Bt, long strideB, int ldb,
                                _Float16* __restrict__ Ct, long strideC, int ldc,
                                int Kdim) {
  int mt = blockIdx.x >> 2;
  int dt = blockIdx.x & 3;
  long hh = (long)blockIdx.z * HH + blockIdx.y;
  const _Float16* Ap = A + strideA * hh + (long)mt * 16 * lda;
  const _Float16* Bp = Bt + strideB * hh + (long)dt * 16 * ldb;
  v8f acc = {};
  for (int k0 = 0; k0 < Kdim; k0 += 32) {
    v16h a = load_frag_a(Ap + k0, lda);
    v16h b = load_frag_b(Bp + k0, ldb);
    acc = wmma32(a, b, acc);
  }
  int lane = threadIdx.x & 31;
  int col  = dt * 16 + (lane & 15);
  int row0 = mt * 16 + 8 * (lane >> 4);
  v8h o;
#pragma unroll
  for (int v = 0; v < 8; ++v) o[v] = (_Float16)acc[v];
  *(v8h*)(Ct + strideC * hh + (long)col * ldc + row0) = o;   // transposed store
}

// ---- fused flash-attention kernel: Yt = (softmax(Qs Ks^T) @ U)^T ------------
// One wave handles 32 query rows (two 16-row WMMA tiles) over the full key
// range, so each K/U B-fragment is consumed by two WMMAs. Softmax without max
// subtraction (S entries ~ unit variance), normalized by exp row-sums at end.
__global__ void flash_dct_kernel(const float* __restrict__ Q,
                                 const _Float16* __restrict__ Kh,
                                 const _Float16* __restrict__ Ut,
                                 _Float16* __restrict__ Yt) {
  int qt = blockIdx.x;                       // 32-row tile index
  long hh = (long)blockIdx.z * HH + blockIdx.y;
  const float* Qp = Q + ((hh * NN) + (long)qt * 32) * DD;
  const _Float16* Kp = Kh + hh * NN * DD;
  const _Float16* Up = Ut + hh * DD * NN;
  _Float16* Yp = Yt + hh * DD * NN;

  int lane = threadIdx.x & 31;
  int r  = lane & 15;
  int kb = (lane >> 4) * 8;
  const float scale = 0.35355339059327373f;

  // Q A-fragments: aq[i][c] = rows [qt*32+i*16, +16), head-dim chunk c*32.
  v16h aq[2][2];
#pragma unroll
  for (int i = 0; i < 2; ++i) {
    const float* qr = Qp + (i * 16 + r) * DD;
#pragma unroll
    for (int j = 0; j < 8; ++j) {
      aq[i][0][j]     = (_Float16)(qr[kb + j] * scale);
      aq[i][0][8 + j] = (_Float16)(qr[16 + kb + j] * scale);
      aq[i][1][j]     = (_Float16)(qr[32 + kb + j] * scale);
      aq[i][1][8 + j] = (_Float16)(qr[48 + kb + j] * scale);
    }
  }

  v8f acc[2][4];
#pragma unroll
  for (int i = 0; i < 2; ++i)
#pragma unroll
    for (int t = 0; t < 4; ++t) acc[i][t] = (v8f){};
  float rsum[2][8];
#pragma unroll
  for (int i = 0; i < 2; ++i)
#pragma unroll
    for (int v = 0; v < 8; ++v) rsum[i][v] = 0.0f;

  __shared__ __align__(16) _Float16 P[32 * 32];  // exp(S) staging, 2 KB

  for (int j0 = 0; j0 < NN; j0 += 32) {
    // Shared K B-fragments: keys [j0,j0+16) and [j0+16,j0+32), d-chunks 0/32.
    v16h bk00 = load_frag_b(Kp + (long)j0 * DD, DD);
    v16h bk01 = load_frag_b(Kp + (long)j0 * DD + 32, DD);
    v16h bk10 = load_frag_b(Kp + (long)(j0 + 16) * DD, DD);
    v16h bk11 = load_frag_b(Kp + (long)(j0 + 16) * DD + 32, DD);

    v8f s[2][2];
#pragma unroll
    for (int i = 0; i < 2; ++i) {
      s[i][0] = (v8f){};
      s[i][0] = wmma32(aq[i][0], bk00, s[i][0]);
      s[i][0] = wmma32(aq[i][1], bk01, s[i][0]);
      s[i][1] = (v8f){};
      s[i][1] = wmma32(aq[i][0], bk10, s[i][1]);
      s[i][1] = wmma32(aq[i][1], bk11, s[i][1]);
    }

    __syncthreads();  // previous iteration's P reads are done
    int c = lane & 15, hf = lane >> 4;
#pragma unroll
    for (int i = 0; i < 2; ++i) {
#pragma unroll
      for (int v = 0; v < 8; ++v) {
        float e0 = __expf(s[i][0][v]);
        float e1 = __expf(s[i][1][v]);
        rsum[i][v] += e0 + e1;
        int row = i * 16 + v + 8 * hf;              // C/D layout row
        P[row * 32 + c]      = (_Float16)e0;
        P[row * 32 + 16 + c] = (_Float16)e1;
      }
    }
    __syncthreads();

    // Re-read P as two 16x32 A-fragments (layout conversion via LDS).
    v16h ap[2];
#pragma unroll
    for (int i = 0; i < 2; ++i) {
      const _Float16* pp = &P[(i * 16 + r) * 32 + kb];
      v8h lo = *(const v8h*)pp;
      v8h hi = *(const v8h*)(pp + 16);
#pragma unroll
      for (int j = 0; j < 8; ++j) { ap[i][j] = lo[j]; ap[i][8 + j] = hi[j]; }
    }
    // Y += P @ U over this 32-key slab; U stored transposed [d, n].
#pragma unroll
    for (int t = 0; t < 4; ++t) {
      v16h bu = load_frag_b(Up + (long)(t * 16) * NN + j0, NN);
      acc[0][t] = wmma32(ap[0], bu, acc[0][t]);
      acc[1][t] = wmma32(ap[1], bu, acc[1][t]);
    }
  }

  // Row sums live spread across the 16 lanes of each half-wave: reduce.
#pragma unroll
  for (int i = 0; i < 2; ++i) {
#pragma unroll
    for (int v = 0; v < 8; ++v) {
      float sv = rsum[i][v];
      sv += __shfl_xor(sv, 1, 32);
      sv += __shfl_xor(sv, 2, 32);
      sv += __shfl_xor(sv, 4, 32);
      sv += __shfl_xor(sv, 8, 32);
      rsum[i][v] = 1.0f / sv;
    }
  }

  // Normalize and store Y transposed [d, n] (contiguous 16B per lane).
  int col0 = lane & 15;
#pragma unroll
  for (int i = 0; i < 2; ++i) {
    int q0 = qt * 32 + i * 16 + 8 * (lane >> 4);
#pragma unroll
    for (int t = 0; t < 4; ++t) {
      v8h o;
#pragma unroll
      for (int v = 0; v < 8; ++v) o[v] = (_Float16)(acc[i][t][v] * rsum[i][v]);
      *(v8h*)(Yp + (long)(t * 16 + col0) * NN + q0) = o;
    }
  }
}

// ---- final GEMM with fp32 row-major output: out = D^T @ Z -------------------
__global__ void gemm_out_kernel(const _Float16* __restrict__ Dth,
                                const _Float16* __restrict__ Zt,
                                float* __restrict__ Out) {
  int qt = blockIdx.x >> 2;
  int dt = blockIdx.x & 3;
  long hh = (long)blockIdx.z * HH + blockIdx.y;
  const _Float16* Ap = Dth + (long)qt * 16 * MM;            // [N, M] row-major
  const _Float16* Bp = Zt + (hh * DD + (long)dt * 16) * MM; // Z^T rows
  v8f acc = {};
  for (int k0 = 0; k0 < MM; k0 += 32) {
    acc = wmma32(load_frag_a(Ap + k0, MM), load_frag_b(Bp + k0, MM), acc);
  }
  float* Op = Out + hh * NN * DD;
  int lane = threadIdx.x & 31;
  int col  = dt * 16 + (lane & 15);
  int row0 = qt * 16 + 8 * (lane >> 4);
#pragma unroll
  for (int v = 0; v < 8; ++v) Op[(long)(row0 + v) * DD + col] = acc[v];
}

// ---- launch -----------------------------------------------------------------

extern "C" void kernel_launch(void* const* d_in, const int* in_sizes, int n_in,
                              void* d_out, int out_size, void* d_ws, size_t ws_size,
                              hipStream_t stream) {
  const float* Q    = (const float*)d_in[0];
  const float* K    = (const float*)d_in[1];
  const float* V    = (const float*)d_in[2];
  const float* mask = (const float*)d_in[3];
  const float* Qdct = (const float*)d_in[4];
  float* Out = (float*)d_out;

  // f16 workspace carve-out (~36 MB total)
  _Float16* ws  = (_Float16*)d_ws;
  const long SZ_HND = (long)BB * HH * NN * DD;  // 4,194,304
  const long SZ_MN  = (long)MM * NN;            //   524,288
  const long SZ_HDM = (long)BB * HH * DD * MM;  //   524,288
  _Float16* Kh  = ws;                 // [B,H,N,D]  K*scale*mask
  _Float16* Vmt = Kh  + SZ_HND;       // [B,H,D,N]  (V*mask)^T
  _Float16* Dh  = Vmt + SZ_HND;       // [M,N]
  _Float16* Dth = Dh  + SZ_MN;        // [N,M]
  _Float16* Vdt = Dth + SZ_MN;        // [B,H,D,M]  (D @ Vm)^T
  _Float16* Ut  = Vdt + SZ_HDM;       // [B,H,D,N]  (D^T @ Vd)^T
  _Float16* Yt  = Ut  + SZ_HND;       // [B,H,D,N]  (A @ U)^T
  _Float16* Zt  = Yt  + SZ_HND;       // [B,H,D,M]  (D @ Y)^T

  prep_kv_kernel<<<(int)(SZ_HND / 256), 256, 0, stream>>>(K, V, mask, Kh, Vmt);
  prep_dct_kernel<<<(int)(SZ_MN / 256), 256, 0, stream>>>(Qdct, Dh, Dth);

  // Vd = D @ Vm           [256 x 64], K = 2048
  gemm_h16_kernel<<<dim3((MM / 16) * 4, HH, BB), 32, 0, stream>>>(
      Dh, 0L, NN, Vmt, (long)DD * NN, NN, Vdt, (long)DD * MM, MM, NN);
  // U = D^T @ Vd          [2048 x 64], K = 256
  gemm_h16_kernel<<<dim3((NN / 16) * 4, HH, BB), 32, 0, stream>>>(
      Dth, 0L, MM, Vdt, (long)DD * MM, MM, Ut, (long)DD * NN, NN, MM);
  // Y = softmax(QsKs^T) @ U  (flash, 32 query rows per wave)
  flash_dct_kernel<<<dim3(NN / 32, HH, BB), 32, 0, stream>>>(Q, Kh, Ut, Yt);
  // Z = D @ Y             [256 x 64], K = 2048
  gemm_h16_kernel<<<dim3((MM / 16) * 4, HH, BB), 32, 0, stream>>>(
      Dh, 0L, NN, Yt, (long)DD * NN, NN, Zt, (long)DD * MM, MM, NN);
  // out = D^T @ Z         [2048 x 64] fp32, K = 256
  gemm_out_kernel<<<dim3((NN / 16) * 4, HH, BB), 32, 0, stream>>>(Dth, Zt, Out);
}
